// VectorQuantizerEMA_66383014527699
// MI455X (gfx1250) — compile-verified, matrix operand order checked by reading, not run on hardware
//
#include <hip/hip_runtime.h>
#include <hip/hip_bf16.h>

typedef __attribute__((ext_vector_type(16))) __bf16 v16bf;
typedef __attribute__((ext_vector_type(8)))  float  v8f;

struct U16x16 { unsigned short v[16]; };

#define HWSZ 4096          // H*W = 64*64
#define NROWS 131072       // B*H*W
#define KCODES 1024
#define DDIM 64
#define QOFF 131072        // quantized offset in d_out
#define POFF (131072 + 8388608)  // perplexity offset in d_out

static __device__ __forceinline__ unsigned short f2bf(float f) {
    unsigned u = __builtin_bit_cast(unsigned, f);
    unsigned r = u + 0x7FFFu + ((u >> 16) & 1u);   // round to nearest even
    return (unsigned short)(r >> 16);
}
static __device__ __forceinline__ float bf2f(unsigned short h) {
    return __builtin_bit_cast(float, ((unsigned)h) << 16);
}

// element e (0..15) of a v16bf operand, lane-group g -> k index inside a 32-wide tile
static __device__ __forceinline__ int kmap(int g, int e) {
    int v = e >> 1, h = e & 1;
    return (v < 4) ? (v * 2 + h + g * 8) : (16 + (v - 4) * 2 + h + g * 8);
}

// ---------------- prep: zero histogram, codebook norms, pack B operands (bf16 hi/lo) ----
__global__ __launch_bounds__(256) void vq_prep(const float* __restrict__ emb,
                                               unsigned short* __restrict__ bhi,
                                               unsigned short* __restrict__ blo,
                                               float* __restrict__ cbh,
                                               int* __restrict__ counts) {
    int t = blockIdx.x * 256 + threadIdx.x;
    if (t < KCODES) {
        counts[t] = 0;
        const float* row = emb + (size_t)t * DDIM;
        float s = 0.f;
        #pragma unroll 8
        for (int d = 0; d < DDIM; ++d) { float f = row[d]; s += f * f; }
        cbh[t] = 0.5f * s;
    } else if (t < KCODES + 2048) {
        int j  = t - KCODES;          // (kb, lane) pack job
        int kb = j >> 5;
        int l  = j & 31;
        int g  = l >> 4;
        int nn = l & 15;
        const float* row = emb + (size_t)(kb * 16 + nn) * DDIM;
        #pragma unroll
        for (int tt = 0; tt < 2; ++tt) {
            #pragma unroll
            for (int e = 0; e < 16; ++e) {
                int d = tt * 32 + kmap(g, e);
                float f = row[d];
                unsigned short hs = f2bf(f);
                size_t o = ((size_t)(kb * 2 + tt) * 32 + l) * 16 + e;
                bhi[o] = hs;
                blo[o] = f2bf(f - bf2f(hs));
            }
        }
    }
}

// ---------------- main: WMMA distance argmin + gather + histogram --------------------
__global__ __launch_bounds__(256) void vq_main(const float* __restrict__ x,
                                               const float* __restrict__ emb,
                                               const unsigned short* __restrict__ bhi,
                                               const unsigned short* __restrict__ blo,
                                               const float* __restrict__ cbh,
                                               int* __restrict__ counts,
                                               float* __restrict__ out) {
    __shared__ int sidx[128];

    const int tid  = threadIdx.x;
    const int wave = tid >> 5;
    const int l    = tid & 31;
    const int g    = l >> 4;
    const int m    = l & 15;
    const int rowbase = blockIdx.x * 128 + wave * 16;

    // ---- build A operand (16 rows x 64 d) in bf16 hi/lo, register resident ----
    {
    }
    const int n  = rowbase + m;
    const int bb = n >> 12;
    const int hh = n & (HWSZ - 1);
    const float* xcol = x + (size_t)bb * DDIM * HWSZ + hh;

    v16bf ahi[2], alo[2];
    #pragma unroll
    for (int tt = 0; tt < 2; ++tt) {
        U16x16 th, tl;
        #pragma unroll
        for (int e = 0; e < 16; ++e) {
            int d = tt * 32 + kmap(g, e);
            float f = xcol[(size_t)d * HWSZ];
            unsigned short hs = f2bf(f);
            th.v[e] = hs;
            tl.v[e] = f2bf(f - bf2f(hs));
        }
        ahi[tt] = __builtin_bit_cast(v16bf, th);
        alo[tt] = __builtin_bit_cast(v16bf, tl);
    }

    float best[8];
    int   bidx[8];
    #pragma unroll
    for (int r = 0; r < 8; ++r) { best[r] = 3.4e38f; bidx[r] = 0; }

    // ---- sweep all 1024 codes, 16 per iteration, 6 WMMAs (bf16x3 split) -------
    for (int kb = 0; kb < 64; ++kb) {
        const v16bf b0h = *(const v16bf*)(bhi + ((size_t)(kb * 2 + 0) * 32 + l) * 16);
        const v16bf b1h = *(const v16bf*)(bhi + ((size_t)(kb * 2 + 1) * 32 + l) * 16);
        const v16bf b0l = *(const v16bf*)(blo + ((size_t)(kb * 2 + 0) * 32 + l) * 16);
        const v16bf b1l = *(const v16bf*)(blo + ((size_t)(kb * 2 + 1) * 32 + l) * 16);

        v8f acc = {};
        acc = __builtin_amdgcn_wmma_f32_16x16x32_bf16(false, ahi[0], false, b0h, (short)0, acc, false, false);
        acc = __builtin_amdgcn_wmma_f32_16x16x32_bf16(false, ahi[1], false, b1h, (short)0, acc, false, false);
        acc = __builtin_amdgcn_wmma_f32_16x16x32_bf16(false, ahi[0], false, b0l, (short)0, acc, false, false);
        acc = __builtin_amdgcn_wmma_f32_16x16x32_bf16(false, ahi[1], false, b1l, (short)0, acc, false, false);
        acc = __builtin_amdgcn_wmma_f32_16x16x32_bf16(false, alo[0], false, b0h, (short)0, acc, false, false);
        acc = __builtin_amdgcn_wmma_f32_16x16x32_bf16(false, alo[1], false, b1h, (short)0, acc, false, false);

        const int   kidx = kb * 16 + m;          // column of this lane (C/D: N = lane%16)
        const float ch   = cbh[kidx];
        #pragma unroll
        for (int r = 0; r < 8; ++r) {
            float s = ch - acc[r];               // ||e||^2/2 - x.e  (argmin-equivalent)
            if (s < best[r]) { best[r] = s; bidx[r] = kidx; }
        }
    }

    // ---- argmin butterfly across the 16-lane halves (wave32) ------------------
    #pragma unroll
    for (int r = 0; r < 8; ++r) {
        #pragma unroll
        for (int off = 1; off < 16; off <<= 1) {
            float ov = __shfl_xor(best[r], off, 32);
            int   oi = __shfl_xor(bidx[r], off, 32);
            if (ov < best[r] || (ov == best[r] && oi < bidx[r])) { best[r] = ov; bidx[r] = oi; }
        }
        if (m == r) sidx[wave * 16 + g * 8 + r] = bidx[r];   // group0->M=r, group1->M=8+r
    }
    __syncthreads();

    // ---- indices + histogram --------------------------------------------------
    if (tid < 128) {
        int idx = sidx[tid];
        atomicAdd(&counts[idx], 1);
        out[blockIdx.x * 128 + tid] = (float)idx;
    }

    // ---- gather quantized rows, write channel-first (B,C,H,W), coalesced ------
    {
        int r      = tid & 127;
        int dstart = tid >> 7;
        int nn2 = blockIdx.x * 128 + r;
        int b2  = nn2 >> 12;
        int h2  = nn2 & (HWSZ - 1);
        const float* erow = emb + (size_t)sidx[r] * DDIM;
        float* oq = out + QOFF + (size_t)b2 * DDIM * HWSZ + h2;
        for (int d = dstart; d < DDIM; d += 2)
            oq[(size_t)d * HWSZ] = erow[d];
    }
}

// ---------------- perplexity ---------------------------------------------------
__global__ __launch_bounds__(1024) void vq_perp(const int* __restrict__ counts,
                                                float* __restrict__ out_p) {
    __shared__ float red[1024];
    int t = threadIdx.x;
    float p = (float)counts[t] * (1.0f / (float)NROWS);
    red[t] = p * logf(p + 1e-10f);
    __syncthreads();
    for (int s = 512; s > 0; s >>= 1) {
        if (t < s) red[t] += red[t + s];
        __syncthreads();
    }
    if (t == 0) *out_p = expf(-red[0]);
}

extern "C" void kernel_launch(void* const* d_in, const int* in_sizes, int n_in,
                              void* d_out, int out_size, void* d_ws, size_t ws_size,
                              hipStream_t stream) {
    const float* x   = (const float*)d_in[0];   // (32,64,64,64) f32
    const float* emb = (const float*)d_in[1];   // (1024,64) f32

    // workspace layout: counts(4KB) | cbh(4KB) | Bhi(128KB) | Blo(128KB)
    int*            counts = (int*)d_ws;
    float*          cbh    = (float*)((char*)d_ws + 4096);
    unsigned short* bhi    = (unsigned short*)((char*)d_ws + 8192);
    unsigned short* blo    = (unsigned short*)((char*)d_ws + 8192 + 131072);

    float* out = (float*)d_out;   // [indices(131072) | quantized(8388608) | perplexity(1)]

    vq_prep<<<12, 256, 0, stream>>>(emb, bhi, blo, cbh, counts);
    vq_main<<<1024, 256, 0, stream>>>(x, emb, bhi, blo, cbh, counts, out);
    vq_perp<<<1, 1024, 0, stream>>>(counts, out + POFF);
}
